// SparseEncoder_15650860827387
// MI455X (gfx1250) — matrix operand order repaired and checked by reading, not compiled, not run on hardware
//
#include <hip/hip_runtime.h>

// ---------------------------------------------------------------------------
// CDNA5 (gfx1250) sparse 3D CNN encoder: implicit-GEMM convs on
// v_wmma_f32_16x16x32_f16 (wave32), f16 activations, f32 accumulation.
// Weight fragments for the two largest-traffic stages are staged into LDS
// with global_load_async_to_lds_b128 (ASYNCcnt) once per block.
// ---------------------------------------------------------------------------

typedef __attribute__((ext_vector_type(16))) _Float16 v16h;
typedef __attribute__((ext_vector_type(8)))  _Float16 v8h;
typedef __attribute__((ext_vector_type(8)))  float    v8f;

typedef __attribute__((address_space(1))) const char gchar;
typedef __attribute__((address_space(3))) char       lchar;

static __device__ __forceinline__ v16h cat16(v8h a, v8h b) {
  return __builtin_shufflevector(a, b, 0,1,2,3,4,5,6,7,8,9,10,11,12,13,14,15);
}

static __device__ __forceinline__ v8f wmma_f16(v16h a, v16h b, v8f c) {
  // D = A(16x32 f16) * B(32x16 f16) + C(16x16 f32)
  return __builtin_amdgcn_wmma_f32_16x16x32_f16(
      /*neg_a=*/false, a, /*neg_b=*/false, b,
      /*c_mod=*/(short)0, c, /*reuse_a=*/false, /*reuse_b=*/false);
}

// ---------------------------------------------------------------------------
// Weight prep: convert [3][3][3][CIN][COUT] f32 into per-(tap,kt,ct) fragment
// blobs of 32 lanes x 16 f16, laid out in this file's contiguous-K B order (A uses the same) for
// v_wmma_*_16x16x32_f16: lane<16 -> col N=lane, K = kt*32 + 0..15;
// lane>=16 -> col N=lane-16, K = kt*32 + 16..31.
// ---------------------------------------------------------------------------
template<int CIN, int COUT>
__global__ void prep_wblob(const float* __restrict__ w, _Float16* __restrict__ blob) {
  constexpr int KT = CIN / 32, CT = COUT / 16;
  int t = blockIdx.x * blockDim.x + threadIdx.x;
  int lane = t & 31, bi = t >> 5;
  if (bi >= 27 * KT * CT) return;
  int ct = bi % CT, kt = (bi / CT) % KT, tap = bi / (CT * KT);
  int kbase = kt * 32 + ((lane & 16) ? 16 : 0);
  int col = ct * 16 + (lane & 15);
  _Float16 tmp[16];
#pragma unroll
  for (int j = 0; j < 16; ++j)
    tmp[j] = (_Float16)w[(size_t)(tap * CIN + kbase + j) * COUT + col];
  v8h* dst = (v8h*)(blob + (size_t)bi * 512 + lane * 16);
  dst[0] = *(v8h*)&tmp[0];
  dst[1] = *(v8h*)&tmp[8];
}

// ---------------------------------------------------------------------------
// Stage 1: Cin=1 -> 32, stride 1, 64^3. K=27 too skinny for WMMA; VALU kernel
// with weights staged in LDS. AdaNorm is per-site over channels -> in-thread.
// ---------------------------------------------------------------------------
__global__ void conv1_kernel(const float* __restrict__ feat,
                             const unsigned char* __restrict__ mask,
                             const float* __restrict__ w1,
                             const float* __restrict__ b1,
                             _Float16* __restrict__ xout) {
  __shared__ float wsm[27 * 32];
  __shared__ float bsm[32];
  for (int i = threadIdx.x; i < 27 * 32; i += blockDim.x) wsm[i] = w1[i];
  if (threadIdx.x < 32) bsm[threadIdx.x] = b1[threadIdx.x];
  __syncthreads();

  const int site = blockIdx.x * blockDim.x + threadIdx.x;  // 4*64^3 exactly
  const int w = site & 63, h = (site >> 6) & 63, d = (site >> 12) & 63, b = site >> 18;

  float acc[32];
#pragma unroll
  for (int c = 0; c < 32; ++c) acc[c] = bsm[c];

  for (int tap = 0; tap < 27; ++tap) {
    const int dz = tap / 9, dy = (tap / 3) % 3, dx = tap % 3;
    const int id = d + dz - 1, ih = h + dy - 1, iw = w + dx - 1;
    float v = 0.f;
    if ((unsigned)id < 64u && (unsigned)ih < 64u && (unsigned)iw < 64u)
      v = feat[(size_t)(((b * 64 + id) * 64 + ih) * 64) + iw];
#pragma unroll
    for (int c = 0; c < 32; ++c) acc[c] = fmaf(v, wsm[tap * 32 + c], acc[c]);
  }

  float mean = 0.f;
#pragma unroll
  for (int c = 0; c < 32; ++c) mean += acc[c];
  mean *= (1.f / 32.f);
  float var = 0.f;
#pragma unroll
  for (int c = 0; c < 32; ++c) { float t = acc[c] - mean; var += t * t; }
  var *= (1.f / 32.f);
  const float rs = rsqrtf(var + 1e-5f);
  const float mk = mask[site] ? 1.f : 0.f;

  _Float16 o[32];
#pragma unroll
  for (int c = 0; c < 32; ++c) {
    float y = (acc[c] - mean) * rs;
    float f = (1.f - 0.1f * y) * y;
    o[c] = (_Float16)(fmaxf(f, 0.f) * mk);
  }
  v8h* dst = (v8h*)(xout + (size_t)site * 32);
#pragma unroll
  for (int q = 0; q < 4; ++q) dst[q] = *(v8h*)&o[q * 8];
}

// ---------------------------------------------------------------------------
// WMMA implicit-GEMM conv. One wave = 16 output sites (consecutive W) x full
// Cout. Loop 27 taps x KT k-chunks, CT WMMAs per chunk. Epilogue transposes
// accumulators through LDS and does per-site AdaNorm+ReLU+mask, f16 store.
// STAGEW: stage the full weight-blob set into LDS via async loads (ASYNCcnt)
// and serve B-fragments from LDS for all taps/waves of the block.
// ---------------------------------------------------------------------------
template<int RI, int RO, int CIN, int COUT, int STRIDE, bool SPARSE, int WPB,
         bool STAGEW>
__global__ void conv_wmma(const _Float16* __restrict__ xin,
                          _Float16* __restrict__ xout,
                          const _Float16* __restrict__ wblob,
                          const float* __restrict__ bias,
                          const unsigned char* __restrict__ mask_in,
                          unsigned char* __restrict__ mask_out) {
  constexpr int KT = CIN / 32, CT = COUT / 16, WT = RO / 16;
  constexpr int WBYTES = 27 * KT * CT * 1024;  // blob-set bytes
  __shared__ float sm[WPB * 16 * COUT];
  __shared__ _Float16 wlds[STAGEW ? (WBYTES / 2) : 8];
  (void)mask_out;

  if constexpr (STAGEW) {
    // Async-copy the whole fragment set (16B chunks) into LDS.
    constexpr int CHUNKS = WBYTES / 16;
    for (int i = threadIdx.x; i < CHUNKS; i += WPB * 32) {
      lchar* l = (lchar*)wlds + i * 16;
      gchar* g = (gchar*)wblob + (size_t)i * 16;
      asm volatile("global_load_async_to_lds_b128 %0, %1, off"
                   :: "v"(l), "v"(g) : "memory");
    }
    asm volatile("s_wait_asynccnt 0" ::: "memory");
    __syncthreads();
  }

  const int wv = threadIdx.x >> 5;         // wave in block (wave32)
  const int lane = threadIdx.x & 31;
  const int m = lane & 15;                 // output site within tile / column
  const int hi8 = (lane & 16) ? 8 : 0;

  int t = blockIdx.x * WPB + wv;
  const int w0 = (t % WT) * 16; t /= WT;
  const int h  = t % RO;        t /= RO;
  const int d  = t % RO;
  const int b  = t / RO;

  v8f zero = {};
  v8f acc[CT];
#pragma unroll
  for (int ct = 0; ct < CT; ++ct) acc[ct] = zero;

  const _Float16* wsrc = STAGEW ? (const _Float16*)wlds : wblob;

  for (int tap = 0; tap < 27; ++tap) {
    const int dz = tap / 9, dy = (tap / 3) % 3, dx = tap % 3;
    const int id = d * STRIDE + dz - 1;
    const int ih = h * STRIDE + dy - 1;
    const int iw = (w0 + m) * STRIDE + dx - 1;
    const bool inb = (unsigned)id < (unsigned)RI &&
                     (unsigned)ih < (unsigned)RI &&
                     (unsigned)iw < (unsigned)RI;
    const _Float16* abase =
        xin + ((size_t)((b * RI + id) * RI + ih) * RI + iw) * CIN;
    if (inb) __builtin_prefetch(abase + (size_t)RI * CIN, 0, 0);  // next-h row hint

#pragma unroll
    for (int kt = 0; kt < KT; ++kt) {
      // A fragment (contiguous-K order, same as B blobs): lane<16 -> K {0..15},
      // lane>=16 -> K {16..31}; channels are contiguous K.
      const int c0 = kt * 32 + (hi8 ? 16 : 0);
      v8h a0 = {}, a1 = {};
      if (inb) {
        a0 = *(const v8h*)(abase + c0);
        a1 = *(const v8h*)(abase + c0 + 8);
      }
      const v16h afrag = cat16(a0, a1);
      const _Float16* bb = wsrc + (size_t)((tap * KT + kt) * CT) * 512 + lane * 16;
#pragma unroll
      for (int ct = 0; ct < CT; ++ct) {
        v8h b0 = *(const v8h*)(bb + (size_t)ct * 512);
        v8h b1 = *(const v8h*)(bb + (size_t)ct * 512 + 8);
        acc[ct] = wmma_f16(afrag, cat16(b0, b1), acc[ct]);
      }
    }
  }

  // Transpose accumulators to (site, channel) through LDS.
  // D layout: lane<16 holds rows 0..7 (vgpr r), lane>=16 rows 8..15, col N=m.
#pragma unroll
  for (int ct = 0; ct < CT; ++ct)
#pragma unroll
    for (int r = 0; r < 8; ++r)
      sm[(wv * 16 + hi8 + r) * COUT + ct * 16 + m] = acc[ct][r];
  asm volatile("s_wait_dscnt 0" ::: "memory");

  if (lane < 16) {
    // AdaNorm per site over channels, then ReLU * mask, store f16.
    float mk;
    if constexpr (SPARSE) {
      bool any = false;
      for (int z = 0; z < 3; ++z)
        for (int y = 0; y < 3; ++y)
          for (int x = 0; x < 3; ++x) {
            int id = d * 2 + z - 1, ih = h * 2 + y - 1, iw = (w0 + m) * 2 + x - 1;
            if ((unsigned)id < (unsigned)RI && (unsigned)ih < (unsigned)RI &&
                (unsigned)iw < (unsigned)RI)
              any |= mask_in[((b * RI + id) * RI + ih) * RI + iw] != 0;
          }
      mask_out[((b * RO + d) * RO + h) * RO + w0 + m] = any ? 1 : 0;
      mk = any ? 1.f : 0.f;
    } else {
      mk = mask_in[((b * RI + d) * RI + h) * RI + (w0 + m)] ? 1.f : 0.f;
    }

    const float* row = &sm[(wv * 16 + m) * COUT];
    float mean = 0.f;
    for (int c = 0; c < COUT; ++c) mean += row[c] + bias[c];
    mean *= (1.f / (float)COUT);
    float var = 0.f;
    for (int c = 0; c < COUT; ++c) {
      float tt = row[c] + bias[c] - mean;
      var += tt * tt;
    }
    var *= (1.f / (float)COUT);
    const float rs = rsqrtf(var + 1e-5f);

    _Float16* dst = xout + ((size_t)((b * RO + d) * RO + h) * RO + w0 + m) * COUT;
    for (int c8 = 0; c8 < COUT; c8 += 8) {
      _Float16 o[8];
#pragma unroll
      for (int j = 0; j < 8; ++j) {
        float y = (row[c8 + j] + bias[c8 + j] - mean) * rs;
        float f = (1.f - 0.1f * y) * y;
        o[j] = (_Float16)(fmaxf(f, 0.f) * mk);
      }
      *(v8h*)(dst + c8) = *(v8h*)o;
    }
  }
}

// ---------------------------------------------------------------------------
// Masked global max-pool over 16^3 + FC 128 -> 256. One block per batch.
// ---------------------------------------------------------------------------
__global__ void pool_fc_kernel(const _Float16* __restrict__ x5,
                               const unsigned char* __restrict__ m3,
                               const float* __restrict__ fcw,
                               const float* __restrict__ fcb,
                               float* __restrict__ out) {
  const int b = blockIdx.x;
  const int tid = threadIdx.x;
  __shared__ float pooled[128];
  if (tid < 128) {
    float mx = -__builtin_inff();
    for (int s = 0; s < 4096; ++s) {
      if (m3[b * 4096 + s]) {
        float v = (float)x5[((size_t)b * 4096 + s) * 128 + tid];
        mx = fmaxf(mx, v);
      }
    }
    pooled[tid] = mx;
  }
  __syncthreads();
  float o = fcb[tid];
  for (int c = 0; c < 128; ++c) o = fmaf(pooled[c], fcw[c * 256 + tid], o);
  out[b * 256 + tid] = o;
}

// ---------------------------------------------------------------------------
// Launcher
// ---------------------------------------------------------------------------
extern "C" void kernel_launch(void* const* d_in, const int* in_sizes, int n_in,
                              void* d_out, int out_size, void* d_ws, size_t ws_size,
                              hipStream_t stream) {
  (void)in_sizes; (void)n_in; (void)out_size; (void)ws_size;
  const float*         feat  = (const float*)d_in[0];
  const unsigned char* mask0 = (const unsigned char*)d_in[1];  // jax bool -> bytes
  const float* w1 = (const float*)d_in[2];  const float* b1 = (const float*)d_in[3];
  const float* w2 = (const float*)d_in[4];  const float* b2 = (const float*)d_in[5];
  const float* w3 = (const float*)d_in[6];  const float* b3 = (const float*)d_in[7];
  const float* w4 = (const float*)d_in[8];  const float* b4 = (const float*)d_in[9];
  const float* w5 = (const float*)d_in[10]; const float* b5 = (const float*)d_in[11];
  const float* fcw = (const float*)d_in[12]; const float* fcb = (const float*)d_in[13];

  char* ws = (char*)d_ws;
  size_t off = 0;
  _Float16* blob2 = (_Float16*)(ws + off); off += (size_t)27 * 1 * 2 * 1024;  //  54 KB
  _Float16* blob3 = (_Float16*)(ws + off); off += (size_t)27 * 1 * 4 * 1024;  // 108 KB
  _Float16* blob4 = (_Float16*)(ws + off); off += (size_t)27 * 2 * 4 * 1024;  // 216 KB
  _Float16* blob5 = (_Float16*)(ws + off); off += (size_t)27 * 2 * 8 * 1024;  // 432 KB
  _Float16* x1 = (_Float16*)(ws + off); off += (size_t)4 * 64 * 64 * 64 * 32 * 2;   // 64 MB
  _Float16* x2 = (_Float16*)(ws + off); off += (size_t)4 * 64 * 64 * 64 * 32 * 2;   // 64 MB
  _Float16* x3 = (_Float16*)(ws + off); off += (size_t)4 * 32 * 32 * 32 * 64 * 2;   // 16 MB
  _Float16* x4 = (_Float16*)(ws + off); off += (size_t)4 * 32 * 32 * 32 * 64 * 2;   // 16 MB
  _Float16* x5 = (_Float16*)(ws + off); off += (size_t)4 * 16 * 16 * 16 * 128 * 2;  //  4 MB
  unsigned char* m2 = (unsigned char*)(ws + off); off += (size_t)4 * 32 * 32 * 32;
  unsigned char* m3 = (unsigned char*)(ws + off); off += (size_t)4 * 16 * 16 * 16;

  // Pack weights into WMMA B-fragment blobs (f32 -> f16).
  prep_wblob<32, 32 ><<<( 54 * 32 + 255) / 256, 256, 0, stream>>>(w2, blob2);
  prep_wblob<32, 64 ><<<(108 * 32 + 255) / 256, 256, 0, stream>>>(w3, blob3);
  prep_wblob<64, 64 ><<<(216 * 32 + 255) / 256, 256, 0, stream>>>(w4, blob4);
  prep_wblob<64, 128><<<(432 * 32 + 255) / 256, 256, 0, stream>>>(w5, blob5);

  // Stage 1: 1->32 @ 64^3 (VALU, K=27 too skinny for WMMA).
  conv1_kernel<<<4096, 256, 0, stream>>>(feat, mask0, w1, b1, x1);

  // Stage 2: subm 32->32 @ 64^3 (dominant stage). Weights staged in LDS (54 KB).
  conv_wmma<64, 64, 32, 32, 1, false, 8, true><<<8192, 256, 0, stream>>>(
      x1, x2, blob2, b2, mask0, nullptr);
  // Stage 3: sparse 32->64, 64^3 -> 32^3. Weights staged in LDS (108 KB).
  conv_wmma<64, 32, 32, 64, 2, true, 8, true><<<1024, 256, 0, stream>>>(
      x2, x3, blob3, b3, mask0, m2);
  // Stage 4: subm 64->64 @ 32^3 (blob 216 KB -> keep global/L2 path).
  conv_wmma<32, 32, 64, 64, 1, false, 8, false><<<1024, 256, 0, stream>>>(
      x3, x4, blob4, b4, m2, nullptr);
  // Stage 5: sparse 64->128, 32^3 -> 16^3 (blob 432 KB > LDS -> global path).
  conv_wmma<32, 16, 64, 128, 2, true, 4, false><<<256, 128, 0, stream>>>(
      x4, x5, blob5, b5, m2, m3);

  // Masked global max pool + FC.
  pool_fc_kernel<<<4, 256, 0, stream>>>(x5, m3, fcw, fcb, (float*)d_out);
}
